// PyTorchBaseline_54322746360202
// MI455X (gfx1250) — matrix operand order repaired and checked by reading, not compiled, not run on hardware
//
#include <hip/hip_runtime.h>
#include <hip/hip_bf16.h>
#include <stdint.h>

// Seq2seq LSTM (T=256,B=128,D=512,H=1024,O=512) for gfx1250.
// bf16 WMMA (v_wmma_f32_16x16x32_bf16) for all GEMMs; f32 cell state.
//  - weights AND activations kept in bf16 memory -> K-loop is pure vmem+wmma
//  - encoder: fused [x_t|h] @ [W_ih_e|W_hh_e]^T, K=1536 (two compile-time loops)
//  - decoder: one-hot input => gather column of W_ih_d; h @ W_hh_d^T via WMMA
//  - FC + softmax + argmax + teacher-forcing select per decode step

constexpr int T_ = 256, B_ = 128, D_ = 512, H_ = 1024, O_ = 512;
constexpr int GH_ = 4 * H_;        // 4096 gate rows
constexpr int KE_ = D_ + H_;       // 1536 fused encoder K

typedef __attribute__((ext_vector_type(16))) __bf16   v16bf;
typedef __attribute__((ext_vector_type(8)))  float    v8f;
typedef __attribute__((ext_vector_type(4)))  unsigned v4u;

union Frag { v16bf v; unsigned short u[16]; v4u q[2]; };

__device__ __forceinline__ unsigned short f2bf(float f) {
  unsigned u = __float_as_uint(f);
  u += 0x7FFFu + ((u >> 16) & 1u);            // round-to-nearest-even
  return (unsigned short)(u >> 16);
}
__device__ __forceinline__ float sigm(float x) { return 1.0f / (1.0f + __expf(-x)); }

// ---------------- packing ----------------
__global__ void pack_enc_kernel(const float* __restrict__ Wih,
                                const float* __restrict__ Whh,
                                unsigned short* __restrict__ dst) {
  int i = blockIdx.x * blockDim.x + threadIdx.x;
  if (i >= GH_ * KE_) return;
  int row = i / KE_, col = i % KE_;
  float v = (col < D_) ? Wih[(size_t)row * D_ + col]
                       : Whh[(size_t)row * H_ + (col - D_)];
  dst[i] = f2bf(v);
}

__global__ void pack_bf16_kernel(const float* __restrict__ src,
                                 unsigned short* __restrict__ dst, int n) {
  int i = blockIdx.x * blockDim.x + threadIdx.x;
  if (i < n) dst[i] = f2bf(src[i]);
}

// one-hot target -> index (argmax over O)
__global__ void tgt_idx_kernel(const float* __restrict__ target, int* __restrict__ tgt) {
  int i = blockIdx.x * blockDim.x + threadIdx.x;
  if (i >= T_ * B_) return;
  const float* p = target + (size_t)i * O_;
  int best = 0; float bv = p[0];
  for (int j = 1; j < O_; ++j) { float v = p[j]; if (v > bv) { bv = v; best = j; } }
  tgt[i] = best;
}

// ---------------- fused LSTM step (encoder & decoder) ----------------
// gates(b, g*H+j) = bias + [x|h](b,:) @ Wbf(g*H+j,:)^T  (+ decoder gather)
// one wave per 16x16 (batch x j) tile, 4 gate accumulators, all-bf16 operands.
// KX = 512 (encoder, fused x part) or 0 (decoder). GATHER = decoder one-hot add.
template <int KX, bool GATHER>
__global__ __launch_bounds__(32)
void lstm_step_kernel(const unsigned short* __restrict__ xbf,    // (B,KX) bf16 or null
                      const unsigned short* __restrict__ hinbf,  // (B,H) bf16
                      const float* __restrict__ cin,             // (B,H) f32
                      const unsigned short* __restrict__ Wbf,    // (4H, KX+H) bf16
                      const float* __restrict__ bias,            // (4H)
                      const float* __restrict__ Wgather,         // W_ih_d f32 or null
                      const int* __restrict__ idx,               // (B) or null
                      unsigned short* __restrict__ houtbf,       // (B,H) bf16
                      float* __restrict__ cout) {
  constexpr int Ktot = KX + H_;
  const int lane = threadIdx.x & 31;
  const int n    = lane & 15;
  const bool hi  = lane >= 16;
  const int b0   = blockIdx.x * 16;
  const int j0   = blockIdx.y * 16;

  v8f acc[4];
#pragma unroll
  for (int g = 0; g < 4; ++g) {
    float bv = bias[g * H_ + j0 + n];
#pragma unroll
    for (int r = 0; r < 8; ++r) acc[g][r] = bv;
  }

  const int arow = b0 + n;
  const unsigned short* wrow = Wbf + (size_t)(j0 + n) * Ktot + (hi ? 8 : 0);

  // ---- x part (encoder only, compile-time) ----
  if (KX > 0) {
    const unsigned short* ap = xbf + (size_t)arow * KX + (hi ? 8 : 0);
    for (int k0 = 0; k0 < KX; k0 += 32) {
      Frag a;
      a.q[0] = ((const v4u*)(ap + k0))[0];        // K = base .. base+7
      a.q[1] = ((const v4u*)(ap + k0 + 16))[0];   // K = base+16 .. base+23
#pragma unroll
      for (int g = 0; g < 4; ++g) {
        const unsigned short* wp = wrow + (size_t)g * H_ * Ktot + k0;
        Frag b;
        b.q[0] = ((const v4u*)wp)[0];
        b.q[1] = ((const v4u*)wp)[2];
        acc[g] = __builtin_amdgcn_wmma_f32_16x16x32_bf16(
            false, a.v, false, b.v, (short)0, acc[g], false, false);
      }
    }
  }

  // ---- h part ----
  {
    const unsigned short* ap = hinbf + (size_t)arow * H_ + (hi ? 8 : 0);
    const unsigned short* wbase = wrow + KX;
    for (int k0 = 0; k0 < H_; k0 += 32) {
      Frag a;
      a.q[0] = ((const v4u*)(ap + k0))[0];
      a.q[1] = ((const v4u*)(ap + k0 + 16))[0];
#pragma unroll
      for (int g = 0; g < 4; ++g) {
        const unsigned short* wp = wbase + (size_t)g * H_ * Ktot + k0;
        Frag b;
        b.q[0] = ((const v4u*)wp)[0];
        b.q[1] = ((const v4u*)wp)[2];
        acc[g] = __builtin_amdgcn_wmma_f32_16x16x32_bf16(
            false, a.v, false, b.v, (short)0, acc[g], false, false);
      }
    }
  }

  // ---- epilogue: LSTM cell ----
#pragma unroll
  for (int r = 0; r < 8; ++r) {
    int brow = b0 + r + (hi ? 8 : 0);
    int j    = j0 + n;
    float iv = acc[0][r], fv = acc[1][r], gv = acc[2][r], ov = acc[3][r];
    if (GATHER) {                           // decoder: one-hot input gather
      int id = idx[brow];
      if (id >= 0) {
        iv += Wgather[(size_t)(0 * H_ + j) * O_ + id];
        fv += Wgather[(size_t)(1 * H_ + j) * O_ + id];
        gv += Wgather[(size_t)(2 * H_ + j) * O_ + id];
        ov += Wgather[(size_t)(3 * H_ + j) * O_ + id];
      }
    }
    float co = cin[(size_t)brow * H_ + j];
    float cn = sigm(fv) * co + sigm(iv) * tanhf(gv);
    float hn = sigm(ov) * tanhf(cn);
    cout[(size_t)brow * H_ + j]   = cn;
    houtbf[(size_t)brow * H_ + j] = f2bf(hn);
  }
}

// ---------------- FC: logits = h @ W_fc^T + b_fc ----------------
__global__ __launch_bounds__(32)
void fc_kernel(const unsigned short* __restrict__ hinbf,   // (B,H) bf16
               const unsigned short* __restrict__ Wbf,      // (O,H) bf16
               const float* __restrict__ bias,
               float* __restrict__ logits) {
  const int lane = threadIdx.x & 31;
  const int n    = lane & 15;
  const bool hi  = lane >= 16;
  const int b0   = blockIdx.x * 16;
  const int o0   = blockIdx.y * 16;

  v8f acc;
  {
    float bv = bias[o0 + n];
#pragma unroll
    for (int r = 0; r < 8; ++r) acc[r] = bv;
  }
  const unsigned short* ap = hinbf + (size_t)(b0 + n) * H_ + (hi ? 8 : 0);
  const unsigned short* wrow = Wbf + (size_t)(o0 + n) * H_ + (hi ? 8 : 0);
  for (int k0 = 0; k0 < H_; k0 += 32) {
    Frag a;
    a.q[0] = ((const v4u*)(ap + k0))[0];
    a.q[1] = ((const v4u*)(ap + k0 + 16))[0];
    Frag b;
    b.q[0] = ((const v4u*)(wrow + k0))[0];
    b.q[1] = ((const v4u*)(wrow + k0))[2];
    acc = __builtin_amdgcn_wmma_f32_16x16x32_bf16(
        false, a.v, false, b.v, (short)0, acc, false, false);
  }
#pragma unroll
  for (int r = 0; r < 8; ++r) {
    int brow = b0 + r + (hi ? 8 : 0);
    logits[(size_t)brow * O_ + o0 + n] = acc[r];
  }
}

// ---------------- softmax + argmax + teacher-forcing select ----------------
__global__ __launch_bounds__(512)
void softmax_select_kernel(const float* __restrict__ logits,
                           const int* __restrict__ tf_mask,  // (T)
                           const int* __restrict__ tgt_idx,  // (T,B)
                           int t,
                           float* __restrict__ out_t,        // (B,O) slice of d_out
                           int* __restrict__ nidx) {         // (B) next input index
  __shared__ float smax[512];
  __shared__ int   sarg[512];
  __shared__ float ssum[512];
  const int b = blockIdx.x;
  const int o = threadIdx.x;
  float v = logits[(size_t)b * O_ + o];
  smax[o] = v; sarg[o] = o;
  __syncthreads();
  for (int s = 256; s > 0; s >>= 1) {
    if (o < s) {
      float v2 = smax[o + s]; int a2 = sarg[o + s];
      if (v2 > smax[o] || (v2 == smax[o] && a2 < sarg[o])) { smax[o] = v2; sarg[o] = a2; }
    }
    __syncthreads();
  }
  float m = smax[0]; int am = sarg[0];
  float e = __expf(v - m);
  ssum[o] = e;
  __syncthreads();
  for (int s = 256; s > 0; s >>= 1) {
    if (o < s) ssum[o] += ssum[o + s];
    __syncthreads();
  }
  out_t[(size_t)b * O_ + o] = e / ssum[0];
  if (o == 0) nidx[b] = (tf_mask[t] > 0) ? tgt_idx[t * B_ + b] : am;
}

// ---------------- host ----------------
extern "C" void kernel_launch(void* const* d_in, const int* in_sizes, int n_in,
                              void* d_out, int out_size, void* d_ws, size_t ws_size,
                              hipStream_t stream) {
  (void)in_sizes; (void)n_in; (void)out_size; (void)ws_size;
  const float* x       = (const float*)d_in[0];
  const float* target  = (const float*)d_in[1];
  const float* h0      = (const float*)d_in[2];
  const float* c0      = (const float*)d_in[3];
  const int*   tf_mask = (const int*)d_in[4];
  const float* Wih_e   = (const float*)d_in[5];
  const float* Whh_e   = (const float*)d_in[6];
  const float* b_e     = (const float*)d_in[7];
  const float* Wih_d   = (const float*)d_in[8];
  const float* Whh_d   = (const float*)d_in[9];
  const float* b_d     = (const float*)d_in[10];
  const float* Wfc     = (const float*)d_in[11];
  const float* b_fc    = (const float*)d_in[12];
  float* out = (float*)d_out;

  // workspace carve-out (~58 MB)
  char* ws = (char*)d_ws;
  size_t off = 0;
  auto carve = [&](size_t bytes) {
    void* p = ws + off;
    off = (off + bytes + 255) & ~(size_t)255;
    return p;
  };
  unsigned short* Wenc_bf = (unsigned short*)carve((size_t)GH_ * KE_ * 2);
  unsigned short* Whhd_bf = (unsigned short*)carve((size_t)GH_ * H_ * 2);
  unsigned short* Wfc_bf  = (unsigned short*)carve((size_t)O_ * H_ * 2);
  unsigned short* xbf     = (unsigned short*)carve((size_t)T_ * B_ * D_ * 2);
  unsigned short* hbuf0   = (unsigned short*)carve((size_t)B_ * H_ * 2);
  unsigned short* hbuf1   = (unsigned short*)carve((size_t)B_ * H_ * 2);
  float* cbuf0 = (float*)carve((size_t)B_ * H_ * 4);
  float* cbuf1 = (float*)carve((size_t)B_ * H_ * 4);
  float* logits = (float*)carve((size_t)B_ * O_ * 4);
  int*   tgtIdx = (int*)carve((size_t)T_ * B_ * 4);
  int*   nidx   = (int*)carve((size_t)B_ * 4);

  // pack weights + activations to bf16, precompute target indices
  pack_enc_kernel<<<(GH_ * KE_ + 255) / 256, 256, 0, stream>>>(Wih_e, Whh_e, Wenc_bf);
  pack_bf16_kernel<<<(GH_ * H_ + 255) / 256, 256, 0, stream>>>(Whh_d, Whhd_bf, GH_ * H_);
  pack_bf16_kernel<<<(O_ * H_ + 255) / 256, 256, 0, stream>>>(Wfc, Wfc_bf, O_ * H_);
  pack_bf16_kernel<<<(T_ * B_ * D_ + 255) / 256, 256, 0, stream>>>(x, xbf, T_ * B_ * D_);
  pack_bf16_kernel<<<(B_ * H_ + 255) / 256, 256, 0, stream>>>(h0, hbuf0, B_ * H_);
  tgt_idx_kernel<<<(T_ * B_ + 255) / 256, 256, 0, stream>>>(target, tgtIdx);

  hipMemcpyAsync(cbuf0, c0, (size_t)B_ * H_ * 4, hipMemcpyDeviceToDevice, stream);

  dim3 gridL(B_ / 16, H_ / 16);   // (8,64)
  dim3 gridF(B_ / 16, O_ / 16);   // (8,32)

  // ---- encoder: 256 sequential fused-GEMM LSTM steps ----
  for (int t = 0; t < T_; ++t) {
    const unsigned short* hi_ = (t & 1) ? hbuf1 : hbuf0;
    unsigned short*       ho_ = (t & 1) ? hbuf0 : hbuf1;
    const float* ci_ = (t & 1) ? cbuf1 : cbuf0;
    float*       co_ = (t & 1) ? cbuf0 : cbuf1;
    lstm_step_kernel<D_, false><<<gridL, 32, 0, stream>>>(
        xbf + (size_t)t * B_ * D_, hi_, ci_, Wenc_bf, b_e,
        nullptr, nullptr, ho_, co_);
  }
  // final h_enc in hbuf0 (t=255 wrote hbuf0); decoder restarts from c0
  hipMemcpyAsync(cbuf0, c0, (size_t)B_ * H_ * 4, hipMemcpyDeviceToDevice, stream);
  hipMemsetAsync(nidx, 0xFF, (size_t)B_ * 4, stream);   // idx=-1 => x_init = 0

  // ---- decoder: 256 autoregressive steps ----
  for (int t = 0; t < T_; ++t) {
    const unsigned short* hi_ = (t & 1) ? hbuf1 : hbuf0;
    unsigned short*       ho_ = (t & 1) ? hbuf0 : hbuf1;
    const float* ci_ = (t & 1) ? cbuf1 : cbuf0;
    float*       co_ = (t & 1) ? cbuf0 : cbuf1;
    lstm_step_kernel<0, true><<<gridL, 32, 0, stream>>>(
        nullptr, hi_, ci_, Whhd_bf, b_d, Wih_d, nidx, ho_, co_);
    fc_kernel<<<gridF, 32, 0, stream>>>(ho_, Wfc_bf, b_fc, logits);
    softmax_select_kernel<<<B_, 512, 0, stream>>>(
        logits, tf_mask, tgtIdx, t, out + (size_t)t * B_ * O_, nidx);
  }
}
